// Med2Vec_22857815950029
// MI455X (gfx1250) — matrix operand order repaired and checked by reading, not compiled, not run on hardware
//
#include <hip/hip_runtime.h>
#include <hip/hip_bf16.h>

typedef __attribute__((ext_vector_type(16))) _Float16 v16h;
typedef __attribute__((ext_vector_type(8)))  _Float16 v8h;
typedef __attribute__((ext_vector_type(8)))  float    v8f;

namespace {
constexpr int   cV  = 30000;
constexpr int   cVp = 30016;   // K-padded (zero-filled) for guard-free tails
constexpr int   cD  = 128;
constexpr int   cN  = 2048;
constexpr int   cE  = 1000000;
constexpr float kEps = 1e-8f;
}

// pack two contiguous 8-half vectors into a WMMA fragment register block
__device__ __forceinline__ v16h cat16(v8h lo, v8h hi) {
  v16h r;
#pragma unroll
  for (int h = 0; h < 8; ++h) { r[h] = lo[h]; r[8 + h] = hi[h]; }
  return r;
}

// ---------------------------------------------------------------- zero accums
__global__ void k_zero(float* acc, float* col_sum) {
  int t = threadIdx.x;
  if (t < 8)  acc[t] = 0.f;
  if (t < cD) col_sum[t] = 0.f;
}

// ---------------- embedding prep: EhT[d][k] f16 (padded), preVec, col_sum ----
__global__ __launch_bounds__(256) void k_prep_emb(const float* __restrict__ emb,
                                                  _Float16* __restrict__ EhT,
                                                  float* __restrict__ preVec,
                                                  float* __restrict__ col_sum) {
  __shared__ float red[256];
  const int tid = threadIdx.x;
  const int d = tid & (cD - 1);  // stride (gridDim*256) is a multiple of 128
  if (blockIdx.x == 0) {         // zero the K padding once
    for (int i = tid; i < cD * (cVp - cV); i += 256)
      EhT[(long long)(i >> 4) * cVp + cV + (i & 15)] = (_Float16)0.f;
  }
  float psum = 0.f;
  const long long total = (long long)cV * cD;
  for (long long idx = (long long)blockIdx.x * 256 + tid; idx < total;
       idx += (long long)gridDim.x * 256) {
    int v = (int)(idx >> 7);
    float e = emb[idx];
    EhT[(long long)d * cVp + v] = (_Float16)e;
    float p = e > 0.f ? e : 0.f;
    preVec[idx] = p;
    psum += p;
  }
  red[tid] = psum;
  __syncthreads();
  if (tid < 128) atomicAdd(&col_sum[d], red[tid] + red[tid + 128]);
}

// ------------------------------------------- W transpose: WhT[v][d] = W[d][v]
__global__ __launch_bounds__(256) void k_prep_w(const float* __restrict__ W,
                                                _Float16* __restrict__ WhT) {
  const long long total = (long long)cD * cV;
  for (long long idx = (long long)blockIdx.x * 256 + threadIdx.x; idx < total;
       idx += (long long)gridDim.x * 256) {
    int v = (int)(idx >> 7), d = (int)(idx & 127);
    WhT[idx] = (_Float16)W[(long long)d * cV + v];  // coalesced store
  }
}

// ------------- GEMM1: visit_rep = relu(x @ E + bias), guard-free WMMA stream
// grid.x = N/16 row blocks; 256 threads = 8 waves; wave w owns cols [16w,16w+16)
__global__ __launch_bounds__(256) void k_gemm1(const float* __restrict__ x,
                                               const _Float16* __restrict__ EhT,
                                               const float* __restrict__ bias,
                                               _Float16* __restrict__ vrh) {
  const int row0 = blockIdx.x * 16;
  const int tid = threadIdx.x, wave = tid >> 5, lane = tid & 31;
  const int g = lane >> 4, mn = lane & 15;
  const int d0 = wave * 16;
  const float*    xrow = x   + (long long)(row0 + mn) * cV;
  const _Float16* brow = EhT + (long long)(d0 + mn) * cVp;

  v8f acc = {};
  int k0 = 0;
  for (; k0 + 32 <= cV; k0 += 32) {  // 937 unguarded iterations
    const float4* xr = (const float4*)(xrow + k0);
    float4 f0 = xr[2 * g], f1 = xr[2 * g + 1];
    float4 f2 = xr[4 + 2 * g], f3 = xr[5 + 2 * g];
    v16h a;
    a[0]  = (_Float16)f0.x; a[1]  = (_Float16)f0.y; a[2]  = (_Float16)f0.z; a[3]  = (_Float16)f0.w;
    a[4]  = (_Float16)f1.x; a[5]  = (_Float16)f1.y; a[6]  = (_Float16)f1.z; a[7]  = (_Float16)f1.w;
    a[8]  = (_Float16)f2.x; a[9]  = (_Float16)f2.y; a[10] = (_Float16)f2.z; a[11] = (_Float16)f2.w;
    a[12] = (_Float16)f3.x; a[13] = (_Float16)f3.y; a[14] = (_Float16)f3.z; a[15] = (_Float16)f3.w;
    v8h blo = *(const v8h*)(brow + k0 + 8 * g);
    v8h bhi = *(const v8h*)(brow + k0 + 16 + 8 * g);
    acc = __builtin_amdgcn_wmma_f32_16x16x32_f16(false, a, false, cat16(blo, bhi),
                                                 (short)0, acc, false, false);
  }
  {  // K tail (k0 = 29984): halves 8..15 of A are beyond V -> compile-time zero;
     // B reads the zero-padded region of EhT. Branch-free.
    const float4* xr = (const float4*)(xrow + k0);
    float4 f0 = xr[2 * g], f1 = xr[2 * g + 1];
    v16h a;
    a[0] = (_Float16)f0.x; a[1] = (_Float16)f0.y; a[2] = (_Float16)f0.z; a[3] = (_Float16)f0.w;
    a[4] = (_Float16)f1.x; a[5] = (_Float16)f1.y; a[6] = (_Float16)f1.z; a[7] = (_Float16)f1.w;
#pragma unroll
    for (int h = 8; h < 16; ++h) a[h] = (_Float16)0.f;
    v8h blo = *(const v8h*)(brow + k0 + 8 * g);
    v8h bhi = *(const v8h*)(brow + k0 + 16 + 8 * g);
    acc = __builtin_amdgcn_wmma_f32_16x16x32_f16(false, a, false, cat16(blo, bhi),
                                                 (short)0, acc, false, false);
  }

  const float bval = bias[d0 + mn];
#pragma unroll
  for (int r = 0; r < 8; ++r) {
    int row = row0 + r + 8 * g;
    float v = acc[r] + bval;
    v = v > 0.f ? v : 0.f;
    vrh[(long long)row * cD + d0 + mn] = (_Float16)v;
  }
}

// ------------------------------------------ norms logits r[v] = preVec . colsum
__global__ __launch_bounds__(256) void k_norms(const float* __restrict__ preVec,
                                               const float* __restrict__ col_sum,
                                               float* __restrict__ rvec) {
  __shared__ float cs[cD];
  if (threadIdx.x < cD) cs[threadIdx.x] = col_sum[threadIdx.x];
  __syncthreads();
  for (int v = blockIdx.x * 256 + threadIdx.x; v < cV; v += gridDim.x * 256) {
    const float* row = preVec + (long long)v * cD;
    float dot = 0.f;
#pragma unroll 8
    for (int d = 0; d < cD; ++d) dot += row[d] * cs[d];
    rvec[v] = dot;
  }
}

// ------- A fragments (K=128) straight from global visit_rep, two b128s per kk
__device__ __forceinline__ void load_afrags(const _Float16* __restrict__ vrh,
                                            int row0, int mn, int g, v16h* af) {
  const _Float16* arow = vrh + (long long)(row0 + mn) * cD;
#pragma unroll
  for (int kk = 0; kk < 4; ++kk) {
    v8h lo = *(const v8h*)(arow + 32 * kk + 16 * g);
    v8h hi = *(const v8h*)(arow + 32 * kk + 16 * g + 8);
    af[kk] = cat16(lo, hi);
  }
}

// 16x16 logit tile: B fragments are contiguous b128 pairs from WhT[v][d]
__device__ __forceinline__ v8f logit_tile(const v16h* af,
                                          const _Float16* __restrict__ WhT,
                                          int c0, int g, int mn) {
  const _Float16* brow = WhT + (long long)(c0 + mn) * cD;
  v8f acc = {};
#pragma unroll
  for (int kk = 0; kk < 4; ++kk) {
    v8h lo = *(const v8h*)(brow + 32 * kk + 16 * g);
    v8h hi = *(const v8h*)(brow + 32 * kk + 16 * g + 8);
    acc = __builtin_amdgcn_wmma_f32_16x16x32_f16(false, af[kk], false, cat16(lo, hi),
                                                 (short)0, acc, false, false);
  }
  return acc;
}

// --------------------- pass A: per-row online softmax stats (m, s) over V
// grid.x = N/16; 256 threads = 8 waves; 1875 16-col tiles split across waves
__global__ __launch_bounds__(256) void k_softmax_stats(
    const _Float16* __restrict__ vrh, const _Float16* __restrict__ WhT,
    const float* __restrict__ bvec, float* __restrict__ mrow,
    float* __restrict__ srow) {
  __shared__ float redm[8][16];
  __shared__ float reds[8][16];
  const int row0 = blockIdx.x * 16;
  const int tid = threadIdx.x, wave = tid >> 5, lane = tid & 31;
  const int g = lane >> 4, mn = lane & 15;

  v16h af[4];
  load_afrags(vrh, row0, mn, g, af);

  float m[8], s[8];
#pragma unroll
  for (int r = 0; r < 8; ++r) { m[r] = -3.0e38f; s[r] = 0.f; }

  for (int t = wave; t < cV / 16; t += 8) {
    v8f acc = logit_tile(af, WhT, t * 16, g, mn);
    const float bb = bvec[t * 16 + mn];
#pragma unroll
    for (int r = 0; r < 8; ++r) {
      float l = acc[r] + bb;
      float nm = fmaxf(m[r], l);
      s[r] = s[r] * __expf(m[r] - nm) + __expf(l - nm);
      m[r] = nm;
    }
  }
  // reduce across the 16 column-lanes of each half (rows r+8g live per half)
#pragma unroll
  for (int off = 1; off < 16; off <<= 1) {
#pragma unroll
    for (int r = 0; r < 8; ++r) {
      float om = __shfl_xor(m[r], off, 32);
      float os = __shfl_xor(s[r], off, 32);
      float nm = fmaxf(m[r], om);
      s[r] = s[r] * __expf(m[r] - nm) + os * __expf(om - nm);
      m[r] = nm;
    }
  }
  if (mn == 0) {
#pragma unroll
    for (int r = 0; r < 8; ++r) {
      redm[wave][r + 8 * g] = m[r];
      reds[wave][r + 8 * g] = s[r];
    }
  }
  __syncthreads();
  if (tid < 16) {
    float M = -3.0e38f, S = 0.f;
#pragma unroll
    for (int w = 0; w < 8; ++w) {
      float om = redm[w][tid], os = reds[w][tid];
      float nm = fmaxf(M, om);
      S = S * __expf(M - nm) + os * __expf(om - nm);
      M = nm;
    }
    mrow[row0 + tid] = M;
    srow[row0 + tid] = S;
  }
}

// ------------------- pass B: recompute logits, pred = softmax, fused CE sums
__global__ __launch_bounds__(256) void k_ce(
    const _Float16* __restrict__ vrh, const _Float16* __restrict__ WhT,
    const float* __restrict__ bvec, const float* __restrict__ mrow,
    const float* __restrict__ srow, const float* __restrict__ x,
    const float* __restrict__ mask, float* __restrict__ acc_ce) {
  __shared__ float red[256];
  const int row0 = blockIdx.x * 16;
  const int tid = threadIdx.x, wave = tid >> 5, lane = tid & 31;
  const int g = lane >> 4, mn = lane & 15;

  v16h af[4];
  load_afrags(vrh, row0, mn, g, af);

  float lm[8], inv_s[8], m1f[8], m1b[8];
#pragma unroll
  for (int r = 0; r < 8; ++r) {
    int gn = row0 + r + 8 * g;
    lm[r] = mrow[gn];
    inv_s[r] = 1.0f / srow[gn];
    float mk = mask[gn];
    m1f[r] = (gn < cN - 1) ? mk * mask[gn + 1] : 0.f;
    m1b[r] = (gn >= 1) ? mask[gn - 1] * mk : 0.f;
  }

  float ce = 0.f;
  for (int t = wave; t < cV / 16; t += 8) {
    const int c0 = t * 16;
    v8f acc = logit_tile(af, WhT, c0, g, mn);  // bit-identical to pass A
    const float bb = bvec[c0 + mn];
    const int c = c0 + mn;
#pragma unroll
    for (int r = 0; r < 8; ++r) {
      int gn = row0 + r + 8 * g;
      float p = __expf((acc[r] + bb) - lm[r]) * inv_s[r];
      if (gn < cN - 1) {  // this pred row as "fwd" vs x[gn+1]
        float fwd = p * m1f[r];
        float xn = x[(long long)(gn + 1) * cV + c];
        ce -= xn * logf(fwd + kEps) + (1.f - xn) * logf(1.f - fwd + kEps);
      }
      if (gn >= 1) {      // this pred row as "bwd" vs x[gn-1]
        float bwd = p * m1b[r];
        float xp = x[(long long)(gn - 1) * cV + c];
        ce -= xp * logf(bwd + kEps) + (1.f - xp) * logf(1.f - bwd + kEps);
      }
    }
  }
  red[tid] = ce;
  __syncthreads();
  for (int off = 128; off > 0; off >>= 1) {
    if (tid < off) red[tid] += red[tid + off];
    __syncthreads();
  }
  if (tid == 0) atomicAdd(acc_ce, red[0]);
}

// ----------------------------------------------- concept-cost pair gather
__global__ __launch_bounds__(256) void k_pairs(const float* __restrict__ preVec,
                                               const float* __restrict__ rvec,
                                               const int* __restrict__ iv,
                                               const int* __restrict__ jv,
                                               float* __restrict__ acc_pair) {
  __shared__ float red[256];
  const int tid = threadIdx.x;
  float local = 0.f;
  for (int e = blockIdx.x * 256 + tid; e < cE; e += gridDim.x * 256) {
    int i = iv[e], j = jv[e];
    const float4* pi = (const float4*)(preVec + (long long)i * cD);
    const float4* pj = (const float4*)(preVec + (long long)j * cD);
    float dot = 0.f;
#pragma unroll 8
    for (int d = 0; d < cD / 4; ++d) {
      float4 a = pi[d], b = pj[d];
      dot += a.x * b.x + a.y * b.y + a.z * b.z + a.w * b.w;
    }
    local += -logf(__expf(dot - rvec[i]) + kEps);
  }
  red[tid] = local;
  __syncthreads();
  for (int off = 128; off > 0; off >>= 1) {
    if (tid < off) red[tid] += red[tid + off];
    __syncthreads();
  }
  if (tid == 0) atomicAdd(acc_pair, red[0]);
}

// ----------------------------------------------------------------- finalize
__global__ void k_final(const float* __restrict__ acc,
                        const float* __restrict__ mask,
                        float* __restrict__ out) {
  float m1 = 0.f;
  for (int n = 0; n < cN - 1; ++n) m1 += mask[n] * mask[n + 1];
  out[0] = acc[0] / (m1 + kEps) + acc[1] / (float)cE;
}

// ---------------------------------------------------------------------------
extern "C" void kernel_launch(void* const* d_in, const int* in_sizes, int n_in,
                              void* d_out, int out_size, void* d_ws,
                              size_t ws_size, hipStream_t stream) {
  const float* x    = (const float*)d_in[0];
  const float* mask = (const float*)d_in[1];
  const float* emb  = (const float*)d_in[2];
  const float* bias = (const float*)d_in[3];
  const float* W    = (const float*)d_in[4];
  const float* bv   = (const float*)d_in[5];
  const int*   iv   = (const int*)d_in[6];
  const int*   jv   = (const int*)d_in[7];
  float* out = (float*)d_out;

  // workspace carve-up (all 256B aligned)
  char* w = (char*)d_ws;
  auto take = [&](size_t bytes) {
    char* p = w;
    w += (bytes + 255) & ~size_t(255);
    return p;
  };
  float*    acc     = (float*)take(8 * 4);                 // [0]=ce [1]=pair
  float*    col_sum = (float*)take(cD * 4);
  float*    mrow    = (float*)take(cN * 4);
  float*    srow    = (float*)take(cN * 4);
  float*    rvec    = (float*)take((size_t)cV * 4);
  float*    preVec  = (float*)take((size_t)cV * cD * 4);
  _Float16* EhT     = (_Float16*)take((size_t)cD * cVp * 2);  // [D][Vpad]
  _Float16* WhT     = (_Float16*)take((size_t)cV * cD * 2);   // [V][D]
  _Float16* vrh     = (_Float16*)take((size_t)cN * cD * 2);   // [N][D]
  (void)ws_size; (void)in_sizes; (void)n_in; (void)out_size;

  k_zero<<<1, 256, 0, stream>>>(acc, col_sum);
  k_prep_emb<<<240, 256, 0, stream>>>(emb, EhT, preVec, col_sum);
  k_prep_w<<<240, 256, 0, stream>>>(W, WhT);
  k_gemm1<<<cN / 16, 256, 0, stream>>>(x, EhT, bias, vrh);
  k_norms<<<120, 256, 0, stream>>>(preVec, col_sum, rvec);
  k_softmax_stats<<<cN / 16, 256, 0, stream>>>(vrh, WhT, bv, mrow, srow);
  k_ce<<<cN / 16, 256, 0, stream>>>(vrh, WhT, bv, mrow, srow, x, mask, &acc[0]);
  k_pairs<<<512, 256, 0, stream>>>(preVec, rvec, iv, jv, &acc[1]);
  k_final<<<1, 1, 0, stream>>>(acc, mask, out);
}
